// TemporalAttentionBlock_68521908240773
// MI455X (gfx1250) — matrix operand order repaired and checked by reading, not compiled, not run on hardware
//
#include <hip/hip_runtime.h>

typedef __bf16 bf16_t;
typedef __attribute__((ext_vector_type(16))) __bf16 v16bf;
typedef __attribute__((ext_vector_type(8)))  __bf16 v8bf;
typedef __attribute__((ext_vector_type(8)))  float  v8f;
typedef int v4i_ __attribute__((vector_size(16)));   // gcc-style, matches builtin param

#define B_   2
#define C_   512
#define F_   32
#define HW_  1024
#define HEADS_ 8
#define DH_  64
#define M_   (B_*HW_*F_)    // 65536 rows
#define NQKV_ (3*C_)        // 1536
#define NEG_MAX (-3.402823466e38f)
#define LDSB_STRIDE 72      // 64 + 8 pad: conflict-minimal for ds_load_b128

#if __has_builtin(__builtin_amdgcn_global_load_async_to_lds_b128)
#define ASYNC_COPY 1
#else
#define ASYNC_COPY 0
#endif

__device__ inline void async_copy_b128(const bf16_t* src, bf16_t* dst) {
#if ASYNC_COPY
  typedef __attribute__((address_space(1))) v4i_ gv4i;
  typedef __attribute__((address_space(3))) v4i_ lv4i;
  __builtin_amdgcn_global_load_async_to_lds_b128((gv4i*)src, (lv4i*)dst, 0, 0);
#else
  *(v8bf*)dst = *(const v8bf*)src;
#endif
}

__device__ inline void wait_async0() {
#if ASYNC_COPY
#if __has_builtin(__builtin_amdgcn_s_wait_asynccnt)
  __builtin_amdgcn_s_wait_asynccnt(0);
#else
  asm volatile("s_wait_asynccnt 0x0" ::: "memory");
#endif
#endif
}

__device__ inline v16bf mk16(v8bf lo, v8bf hi) {
  v16bf r;
#pragma unroll
  for (int i = 0; i < 8; ++i) { r[i] = lo[i]; r[i + 8] = hi[i]; }
  return r;
}

// ---------------- Kernel 1: GroupNorm statistics ----------------
__global__ void gn_stats(const float* __restrict__ x, float* __restrict__ stats) {
  __shared__ float s1[256], s2[256];
  const int bg = blockIdx.x;                       // 0..63
  const float4* p = (const float4*)(x + (size_t)bg * 524288);
  float s = 0.f, ss = 0.f;
  for (int i = threadIdx.x; i < 131072; i += 256) {
    float4 v = p[i];
    s  += v.x + v.y + v.z + v.w;
    ss += v.x*v.x + v.y*v.y + v.z*v.z + v.w*v.w;
  }
  s1[threadIdx.x] = s; s2[threadIdx.x] = ss;
  __syncthreads();
  for (int off = 128; off > 0; off >>= 1) {
    if (threadIdx.x < off) {
      s1[threadIdx.x] += s1[threadIdx.x + off];
      s2[threadIdx.x] += s2[threadIdx.x + off];
    }
    __syncthreads();
  }
  if (threadIdx.x == 0) {
    float mean = s1[0] * (1.f / 524288.f);
    float var  = s2[0] * (1.f / 524288.f) - mean * mean;
    stats[bg * 2]     = mean;
    stats[bg * 2 + 1] = rsqrtf(var + 1e-5f);
  }
}

// ------- Kernel 2: normalize + affine + transpose to [b,hw,f,c] bf16 -------
__global__ void gn_apply_transpose(const float* __restrict__ x,
                                   const float* __restrict__ stats,
                                   const float* __restrict__ nw,
                                   const float* __restrict__ nb,
                                   bf16_t* __restrict__ t) {
  __shared__ __align__(16) float tile[32][33];
  int id = blockIdx.x;
  const int ct = id & 15;  id >>= 4;               // C/32 = 16 tiles
  const int h  = id & 31;  id >>= 5;
  const int f  = id & 31;  id >>= 5;
  const int b  = id;
  const int tx = threadIdx.x, ty = threadIdx.y;
  const int c0 = ct * 32;
#pragma unroll
  for (int cc = ty; cc < 32; cc += 8) {
    const int c = c0 + cc;
    const float mean = stats[(b * 32 + (c >> 4)) * 2];
    const float rstd = stats[(b * 32 + (c >> 4)) * 2 + 1];
    const float v = x[(((size_t)(b * C_ + c) * F_ + f) << 10) + (h << 5) + tx];
    tile[cc][tx] = (v - mean) * rstd * nw[c] + nb[c];
  }
  __syncthreads();
#pragma unroll
  for (int ww = ty; ww < 32; ww += 8) {
    const int n = (h << 5) + ww;                   // hw index
    t[(((size_t)(b << 10) + n) * F_ + f) * C_ + c0 + tx] = (bf16_t)tile[tx][ww];
  }
}

// ------- Kernel 3: weight transpose  w[R][Ccols] -> wt[Ccols][R] (bf16) -------
__global__ void transpose_w_bf16(const float* __restrict__ w, bf16_t* __restrict__ wt,
                                 int R, int Ccols) {
  const int idx = blockIdx.x * blockDim.x + threadIdx.x;
  if (idx >= R * Ccols) return;
  const int k = idx / R;
  const int n = idx - k * R;
  wt[(size_t)k * R + n] = (bf16_t)w[(size_t)n * Ccols + k];
}

// -------- shared GEMM mainloop: 32(M) x 64(N) per wave, B panel via LDS --------
__device__ inline void gemm_mainloop(const bf16_t* __restrict__ A,
                                     const bf16_t* __restrict__ BT,
                                     int N, int K, int m0, int n0,
                                     int lane, bf16_t* ldsB, v8f acc[8]) {
  const int kbase = (lane < 16) ? 0 : 8;           // 16-bit A-matrix VGPR layout
  const bf16_t* arow0 = A + (size_t)(m0 + (lane & 15)) * K;
  const bf16_t* arow1 = arow0 + ((size_t)K << 4);
  const int br = threadIdx.x >> 3;                 // 0..31 : panel row
  const int bc = (threadIdx.x & 7) << 3;           // 0..56 : 8-elem col chunk
  for (int k0 = 0; k0 < K; k0 += 32) {
    __syncthreads();                               // previous panel consumed
    async_copy_b128(BT + (size_t)(k0 + br) * N + n0 + bc,
                    ldsB + br * LDSB_STRIDE + bc);
    wait_async0();
    __syncthreads();                               // panel visible to all waves

    __builtin_prefetch(arow0 + k0 + 64, 0, 1);
    v8bf a0lo = *(const v8bf*)(arow0 + k0 + kbase);
    v8bf a0hi = *(const v8bf*)(arow0 + k0 + 16 + kbase);
    v8bf a1lo = *(const v8bf*)(arow1 + k0 + kbase);
    v8bf a1hi = *(const v8bf*)(arow1 + k0 + 16 + kbase);
    const v16bf a0 = mk16(a0lo, a0hi);
    const v16bf a1 = mk16(a1lo, a1hi);
    const bf16_t* bcol = ldsB + (lane & 15);  // lane%16 == N column of B
#pragma unroll
    for (int tn = 0; tn < 4; ++tn) {
      v16bf bf;
#pragma unroll
      for (int s = 0; s < 16; ++s)
        bf[s] = bcol[(((s & 8) << 1) + kbase + (s & 7)) * LDSB_STRIDE + tn * 16];
      acc[tn]     = __builtin_amdgcn_wmma_f32_16x16x32_bf16(false, a0, false, bf,
                                                            (short)0, acc[tn], false, false);
      acc[4 + tn] = __builtin_amdgcn_wmma_f32_16x16x32_bf16(false, a1, false, bf,
                                                            (short)0, acc[4 + tn], false, false);
    }
  }
}

// ------------- Kernel 4: QKV GEMM (bf16 out + bias) -------------
__global__ void gemm_bf16_bias(const bf16_t* __restrict__ A, const bf16_t* __restrict__ BT,
                               const float* __restrict__ bias, bf16_t* __restrict__ Out,
                               int M, int N, int K) {
  __shared__ __align__(16) bf16_t ldsB[32 * LDSB_STRIDE];
  const int lane = threadIdx.x & 31;
  const int wave = threadIdx.x >> 5;
  const int nPan = N >> 6;
  const int bm = blockIdx.x / nPan;
  const int bn = blockIdx.x - bm * nPan;
  const int m0 = (bm << 8) + (wave << 5);
  const int n0 = bn << 6;
  v8f acc[8];
#pragma unroll
  for (int t = 0; t < 8; ++t) acc[t] = (v8f){0.f,0.f,0.f,0.f,0.f,0.f,0.f,0.f};
  gemm_mainloop(A, BT, N, K, m0, n0, lane, ldsB, acc);

  const int lr = lane & 15;
  const int rsel = (lane < 16) ? 0 : 8;            // f32 C/D layout
#pragma unroll
  for (int tn = 0; tn < 4; ++tn) {
    const int col = n0 + tn * 16 + lr;
    const float bv = bias[col];
#pragma unroll
    for (int half = 0; half < 2; ++half) {
      const int rbase = m0 + half * 16 + rsel;
#pragma unroll
      for (int r = 0; r < 8; ++r)
        Out[(size_t)(rbase + r) * N + col] = (bf16_t)(acc[half * 4 + tn][r] + bv);
    }
  }
}

// ---------------- Kernel 5: temporal attention ----------------
__global__ void attn_kernel(const bf16_t* __restrict__ QKV,
                            const float* __restrict__ pos_bias,
                            const unsigned char* __restrict__ vmask,
                            bf16_t* __restrict__ Obf) {
  __shared__ __align__(16) float  simLds[HEADS_][32][33];
  __shared__ __align__(16) bf16_t attnLds[HEADS_][32][32];
  const int head = threadIdx.x >> 5;
  const int lane = threadIdx.x & 31;
  const int bn = blockIdx.x;                       // b*1024 + hw
  const int b = bn >> 10;
  const size_t mbase = (size_t)bn * F_;
  const float scale = 0.125f;                      // dh^-0.5

  const int kb = (lane < 16) ? 0 : 8;
#pragma unroll
  for (int ti = 0; ti < 2; ++ti) {
    const int i0 = ti << 4;
    const bf16_t* qrow = QKV + (mbase + i0 + (lane & 15)) * NQKV_ + (size_t)head * DH_;
#pragma unroll
    for (int tj = 0; tj < 2; ++tj) {
      const int j0 = tj << 4;
      v8f acc = {0.f,0.f,0.f,0.f,0.f,0.f,0.f,0.f};
#pragma unroll
      for (int kk = 0; kk < DH_; kk += 32) {
        v8bf alo = *(const v8bf*)(qrow + kk + kb);
        v8bf ahi = *(const v8bf*)(qrow + kk + 16 + kb);
        const bf16_t* krow = QKV + (mbase + j0 + (lane & 15)) * NQKV_ + 512 + (size_t)head * DH_;  // B = k^T, lane%16 = j
        const v16bf bfr = mk16(*(const v8bf*)(krow + kk + kb), *(const v8bf*)(krow + kk + 16 + kb));
        acc = __builtin_amdgcn_wmma_f32_16x16x32_bf16(false, mk16(alo, ahi),
                                                      false, bfr,
                                                      (short)0, acc, false, false);
      }
      const int j = j0 + (lane & 15);
      const int ib = i0 + ((lane < 16) ? 0 : 8);
      const bool mj = vmask[b * F_ + j] != 0;
#pragma unroll
      for (int r = 0; r < 8; ++r) {
        const int i = ib + r;
        const bool mi = vmask[b * F_ + i] != 0;
        float v = acc[r] * scale + pos_bias[((size_t)head * F_ + i) * F_ + j];
        simLds[head][i][j] = (mi && mj) ? v : NEG_MAX;
      }
    }
  }
  __syncthreads();

  float mx = NEG_MAX;
#pragma unroll
  for (int j = 0; j < 32; ++j) mx = fmaxf(mx, simLds[head][lane][j]);
  float sum = 0.f;
#pragma unroll
  for (int j = 0; j < 32; ++j) {
    float e = __expf(simLds[head][lane][j] - mx);
    sum += e;
    simLds[head][lane][j] = e;
  }
  const float inv = 1.f / sum;
#pragma unroll
  for (int j = 0; j < 32; ++j)
    attnLds[head][lane][j] = (bf16_t)(simLds[head][lane][j] * inv);
  __syncthreads();

#pragma unroll
  for (int ti = 0; ti < 2; ++ti) {
    const int i0 = ti << 4;
    const bf16_t* arow = &attnLds[head][i0 + (lane & 15)][0];
    v8bf alo = *(const v8bf*)(arow + kb);
    v8bf ahi = *(const v8bf*)(arow + 16 + kb);
    const v16bf afr = mk16(alo, ahi);
#pragma unroll
    for (int td = 0; td < 4; ++td) {
      const int d0 = td << 4;
      const bf16_t* vcol = QKV + mbase * NQKV_ + 1024 + (size_t)head * DH_ + d0 + (lane & 15);
      v16bf bv;
#pragma unroll
      for (int s = 0; s < 16; ++s)
        bv[s] = vcol[(size_t)(((s & 8) << 1) + kb + (s & 7)) * NQKV_];
      v8f acc = {0.f,0.f,0.f,0.f,0.f,0.f,0.f,0.f};
      acc = __builtin_amdgcn_wmma_f32_16x16x32_bf16(false, afr, false, bv,
                                                    (short)0, acc, false, false);
      const int col = head * DH_ + d0 + (lane & 15);
      const int ib = i0 + ((lane < 16) ? 0 : 8);
#pragma unroll
      for (int r = 0; r < 8; ++r)
        Obf[(mbase + ib + r) * C_ + col] = (bf16_t)acc[r];
    }
  }
}

// ---- Kernel 6: out projection + bias + residual, scattered to [b,c,f,h,w] ----
__global__ void gemm_out(const bf16_t* __restrict__ A, const bf16_t* __restrict__ BT,
                         const float* __restrict__ bias, const float* __restrict__ x,
                         float* __restrict__ out) {
  __shared__ __align__(16) bf16_t ldsB[32 * LDSB_STRIDE];
  const int lane = threadIdx.x & 31;
  const int wave = threadIdx.x >> 5;
  const int nPan = C_ >> 6;                        // 8
  const int bm = blockIdx.x / nPan;
  const int bn = blockIdx.x - bm * nPan;
  const int m0 = (bm << 8) + (wave << 5);
  const int n0 = bn << 6;
  v8f acc[8];
#pragma unroll
  for (int t = 0; t < 8; ++t) acc[t] = (v8f){0.f,0.f,0.f,0.f,0.f,0.f,0.f,0.f};
  gemm_mainloop(A, BT, C_, C_, m0, n0, lane, ldsB, acc);

  const int lr = lane & 15;
  const int rsel = (lane < 16) ? 0 : 8;
#pragma unroll
  for (int tn = 0; tn < 4; ++tn) {
    const int col = n0 + tn * 16 + lr;             // channel c
    const float bv = bias[col];
#pragma unroll
    for (int half = 0; half < 2; ++half) {
      const int rbase = m0 + half * 16 + rsel;
#pragma unroll
      for (int r = 0; r < 8; ++r) {
        const int m = rbase + r;                   // m = (b*1024 + hw)*32 + f
        const int f  = m & 31;
        const int hw = (m >> 5) & 1023;
        const int b  = m >> 15;
        const size_t oi = (((size_t)(b * C_ + col) * F_ + f) << 10) + hw;
        out[oi] = x[oi] + acc[half * 4 + tn][r] + bv;
      }
    }
  }
}

extern "C" void kernel_launch(void* const* d_in, const int* in_sizes, int n_in,
                              void* d_out, int out_size, void* d_ws, size_t ws_size,
                              hipStream_t stream) {
  const float* x        = (const float*)d_in[0];
  const float* pos_bias = (const float*)d_in[1];
  const unsigned char* vmask = (const unsigned char*)d_in[2];
  const float* norm_w   = (const float*)d_in[3];
  const float* norm_b   = (const float*)d_in[4];
  const float* qkv_w    = (const float*)d_in[5];
  const float* qkv_b    = (const float*)d_in[6];
  const float* out_w    = (const float*)d_in[7];
  const float* out_b    = (const float*)d_in[8];
  float* out = (float*)d_out;

  char* ws = (char*)d_ws;
  float*  stats  = (float*)ws;                                    //   512 B
  bf16_t* t_bf   = (bf16_t*)(ws + 1024);                          //  64 MB
  bf16_t* qkv_wT = (bf16_t*)(ws + 1024 + 67108864ULL);            // 1.5 MB
  bf16_t* out_wT = (bf16_t*)(ws + 1024 + 67108864ULL + 1572864ULL);
  bf16_t* qkv_o  = (bf16_t*)(ws + 1024 + 67108864ULL + 1572864ULL + 524288ULL); // 192 MB
  bf16_t* o_bf   = t_bf;   // t is dead after the QKV GEMM; reuse its space

  gn_stats<<<64, 256, 0, stream>>>(x, stats);
  dim3 tblk(32, 8);
  gn_apply_transpose<<<32768, tblk, 0, stream>>>(x, stats, norm_w, norm_b, t_bf);
  transpose_w_bf16<<<(NQKV_ * C_ + 255) / 256, 256, 0, stream>>>(qkv_w, qkv_wT, NQKV_, C_);
  transpose_w_bf16<<<(C_ * C_ + 255) / 256, 256, 0, stream>>>(out_w, out_wT, C_, C_);
  // QKV: (65536/256) m-blocks x (1536/64) n-panels = 256 x 24 = 6144 blocks
  gemm_bf16_bias<<<6144, 256, 0, stream>>>(t_bf, qkv_wT, qkv_b, qkv_o, M_, NQKV_, C_);
  attn_kernel<<<B_ * HW_, 256, 0, stream>>>(qkv_o, pos_bias, vmask, o_bf);
  // OUT: (65536/256) x (512/64) = 256 x 8 = 2048 blocks
  gemm_out<<<2048, 256, 0, stream>>>(o_bf, out_wT, out_b, x, out);
}
